// CustomModel_71674414235775
// MI455X (gfx1250) — compile-verified, matrix operand order checked by reading, not run on hardware
//
#include <hip/hip_runtime.h>

// ---------------------------------------------------------------------------
// Types for CDNA5 WMMA / TDM
// ---------------------------------------------------------------------------
typedef __bf16 bf16;
typedef __attribute__((ext_vector_type(16))) __bf16 v16bf;
typedef __attribute__((ext_vector_type(8)))  __bf16 v8bf;
typedef __attribute__((ext_vector_type(8)))  float  v8f;
typedef __attribute__((ext_vector_type(4)))  float  v4f;
typedef __attribute__((ext_vector_type(4)))  unsigned int v4u;
typedef __attribute__((ext_vector_type(8)))  int v8i;
typedef __attribute__((ext_vector_type(4)))  int v4i;

#define B_DIM 4096
#define D_DIM 768
#define F_DIM 6144
#define C_DIM 1000
#define K_TOP 64

#define TILE 128
#define TK   32
#define LDA  40   // padded LDS row (bf16 elems): 80B stride -> conflict-free b128 reads

#if __has_builtin(__builtin_amdgcn_tensor_load_to_lds) && \
    __has_builtin(__builtin_amdgcn_s_wait_tensorcnt)
#define USE_TDM 1
#else
#define USE_TDM 0
#endif

__device__ __forceinline__ bf16 f2bf(float f) { return (bf16)f; }

// ---------------------------------------------------------------------------
// Kernel 1: h = relu(x @ W_inter + b_inter) via v_wmma_f32_16x16x32_bf16
//   grid (48, 32), block 256 (8 wave32s). Each wave: 64x32 out = 4x2 WMMA tiles.
// ---------------------------------------------------------------------------
__global__ __launch_bounds__(256) void gemm1_bias_relu(
    const float* __restrict__ x,      // [4096, 768]
    const float* __restrict__ W,      // [768, 6144]
    const float* __restrict__ bias,   // [6144]
    float* __restrict__ h)            // [4096, 6144]
{
    __shared__ bf16 As[TILE][LDA];    // M-major, K contiguous
    __shared__ bf16 Bs[TILE][LDA];    // N-major (transposed), K contiguous

    const int t    = threadIdx.x;
    const int lane = t & 31;
    const int wid  = t >> 5;
    const int wm   = (wid & 1) * 64;  // wave M offset within tile
    const int wn   = (wid >> 1) * 32; // wave N offset within tile
    const int tile_m = blockIdx.y * TILE;
    const int tile_n = blockIdx.x * TILE;

    const int kbase = (lane & 16) ? 8 : 0;  // 16-bit A/B fragment K sub-base
    const int lrow  = lane & 15;

    v8f acc[4][2];
    #pragma unroll
    for (int mi = 0; mi < 4; ++mi)
        #pragma unroll
        for (int ni = 0; ni < 2; ++ni)
            acc[mi][ni] = (v8f){0.f,0.f,0.f,0.f,0.f,0.f,0.f,0.f};

    for (int k0 = 0; k0 < D_DIM; k0 += TK) {
        // ---- stage A tile: x[tile_m..+128][k0..+32] -> bf16 LDS -------------
        #pragma unroll
        for (int i = 0; i < 4; ++i) {
            int idx = t + i * 256;              // 1024 float4 slots
            int r   = idx >> 3;                 // 8 float4 per row
            int kk  = (idx & 7) * 4;
            v4f a4 = *(const v4f*)(x + (size_t)(tile_m + r) * D_DIM + k0 + kk);
            As[r][kk + 0] = f2bf(a4[0]);
            As[r][kk + 1] = f2bf(a4[1]);
            As[r][kk + 2] = f2bf(a4[2]);
            As[r][kk + 3] = f2bf(a4[3]);
        }
        // ---- stage B tile: W[k0..+32][tile_n..+128] -> transposed bf16 LDS --
        #pragma unroll
        for (int i = 0; i < 4; ++i) {
            int idx = t + i * 256;              // 1024 float4 slots
            int kk  = idx >> 5;                 // 32 float4 per K-row
            int c4  = (idx & 31) * 4;
            v4f b4 = *(const v4f*)(W + (size_t)(k0 + kk) * F_DIM + tile_n + c4);
            Bs[c4 + 0][kk] = f2bf(b4[0]);
            Bs[c4 + 1][kk] = f2bf(b4[1]);
            Bs[c4 + 2][kk] = f2bf(b4[2]);
            Bs[c4 + 3][kk] = f2bf(b4[3]);
        }
        __syncthreads();

        // Prefetch next K tile into cache while we compute (global_prefetch_b8)
        if (k0 + TK < D_DIM) {
            __builtin_prefetch(x + (size_t)(tile_m + (t >> 1)) * D_DIM + (k0 + TK) + (t & 1) * 16, 0, 1);
            __builtin_prefetch(W + (size_t)(k0 + TK + (t >> 3)) * F_DIM + tile_n + (t & 7) * 16, 0, 1);
        }

        // ---- fragments + 8 WMMAs -------------------------------------------
        v16bf afrag[4], bfrag[2];
        #pragma unroll
        for (int mi = 0; mi < 4; ++mi) {
            const bf16* p = &As[wm + mi * 16 + lrow][kbase];
            union { v16bf v; v8bf half[2]; } u;
            u.half[0] = *(const v8bf*)(p);        // K = kbase .. kbase+7
            u.half[1] = *(const v8bf*)(p + 16);   // K = kbase+16 .. kbase+23
            afrag[mi] = u.v;
        }
        #pragma unroll
        for (int ni = 0; ni < 2; ++ni) {
            const bf16* p = &Bs[wn + ni * 16 + lrow][kbase];
            union { v16bf v; v8bf half[2]; } u;
            u.half[0] = *(const v8bf*)(p);
            u.half[1] = *(const v8bf*)(p + 16);
            bfrag[ni] = u.v;
        }
        #pragma unroll
        for (int mi = 0; mi < 4; ++mi)
            #pragma unroll
            for (int ni = 0; ni < 2; ++ni)
                acc[mi][ni] = __builtin_amdgcn_wmma_f32_16x16x32_bf16(
                    false, afrag[mi], false, bfrag[ni],
                    (short)0, acc[mi][ni], false, false);
        __syncthreads();
    }

    // ---- epilogue: bias + relu, write dense h ------------------------------
    const int mhi = (lane & 16) ? 8 : 0;
    #pragma unroll
    for (int ni = 0; ni < 2; ++ni) {
        const int col = tile_n + wn + ni * 16 + lrow;
        const float bb = bias[col];
        #pragma unroll
        for (int mi = 0; mi < 4; ++mi) {
            #pragma unroll
            for (int r = 0; r < 8; ++r) {
                int row = tile_m + wm + mi * 16 + mhi + r;
                float v = acc[mi][ni][r] + bb;
                h[(size_t)row * F_DIM + col] = v > 0.f ? v : 0.f;
            }
        }
    }
}

// ---------------------------------------------------------------------------
// Kernel 2: per-row exact top-64 (radix select on float bits; valid post-ReLU),
// in-place scatter-to-sparse, then sparse row @ W_cls (+ b_cls).
//
// Data movement reasoning for MI455X:
//  * The 24 KB h-row is fetched with ONE Tensor-Data-Mover descriptor load
//    into LDS (TENSORcnt), instead of 24 strided per-thread global loads.
//  * W_cls (24.6 MB) stays resident in the 192 MB L2 across all 4096 rows, so
//    the 64-row gathers (1 GB of logical traffic) are L2 hits, not HBM.
//  * sparse/logits outputs are written once and never re-read on device ->
//    non-temporal stores so they don't evict W_cls / h from L2.
//   grid 4096, block 256. Each thread owns 24 of the 6144 features.
// ---------------------------------------------------------------------------
__global__ __launch_bounds__(256) void topk_scatter_cls(
    float* __restrict__ hs,          // [4096, 6144] in: dense h / out: sparse
    const float* __restrict__ Wc,    // [6144, 1000]
    const float* __restrict__ bc,    // [1000]
    float* __restrict__ logits)      // [4096, 1000]
{
    const int row = blockIdx.x;
    const int t   = threadIdx.x;
    float* hrow = hs + (size_t)row * F_DIM;

    __shared__ float    hbuf[F_DIM];    // 24 KB row staging
    __shared__ unsigned hist[256];
    __shared__ unsigned sh_sel;
    __shared__ int      sh_kk;
    __shared__ int      nsel, neq;
    __shared__ float    tval[80];
    __shared__ int      tidx[80];

#if USE_TDM
    // ---- TDM: one descriptor load pulls the whole row into LDS -------------
    if (t < 32) {   // wave 0 issues the tensor DMA (wave-level op, uniform D#)
        unsigned long long ga = (unsigned long long)(const void*)hrow;
        unsigned lds_byte = (unsigned)(unsigned long long)(const void*)&hbuf[0];

        v4u g0;
        g0[0] = 1u;                                   // count=1, user descriptor
        g0[1] = lds_byte;                             // lds_addr (bytes)
        g0[2] = (unsigned)(ga & 0xFFFFFFFFu);         // global_addr[31:0]
        g0[3] = (unsigned)((ga >> 32) & 0x01FFFFFFu)  // global_addr[56:32]
              | (2u << 30);                           // type = 2 ("image")

        v8i g1;
        g1[0] = (int)(2u << 16);                      // wg_mask=0, data_size=4B
        g1[1] = (int)((unsigned)(F_DIM & 0xFFFF) << 16); // tensor_dim0[15:0]
        g1[2] = (int)(((unsigned)F_DIM >> 16)         // tensor_dim0[31:16]
              | (1u << 16));                          // tensor_dim1 = 1
        g1[3] = (int)((unsigned)F_DIM << 16);         // tile_dim0 = 6144
        g1[4] = (int)1;                               // tile_dim1 = 1, tile_dim2 = 0
        g1[5] = (int)F_DIM;                           // tensor_dim0_stride[31:0]
        g1[6] = 0;                                    // stride hi / dim1_stride lo
        g1[7] = 0;

        v4i gz  = {0, 0, 0, 0};                       // groups 2/3 unused (<=2D)
        v8i gz8 = {0, 0, 0, 0, 0, 0, 0, 0};           // 6-arg toolchain extra group
        __builtin_amdgcn_tensor_load_to_lds(g0, g1, gz, gz, gz8, 0);
        __builtin_amdgcn_s_wait_tensorcnt(0);
    }
    __syncthreads();
#else
    // Fallback: cooperative staging through LDS
    #pragma unroll
    for (int j = 0; j < 24; ++j) hbuf[t + j * 256] = hrow[t + j * 256];
    __syncthreads();
#endif

    float    v[24];
    unsigned key[24];
    #pragma unroll
    for (int j = 0; j < 24; ++j) {
        float f = hbuf[t + j * 256];
        v[j]   = f;
        key[j] = __float_as_uint(f);   // >=0 after relu -> uint order == float order
    }

    // --- 4-pass MSB-first radix select for the 64th-largest key -------------
    unsigned prefix = 0, pmask = 0;
    int kk = K_TOP;
    for (int p = 3; p >= 0; --p) {
        hist[t] = 0;
        __syncthreads();
        #pragma unroll
        for (int j = 0; j < 24; ++j)
            if ((key[j] & pmask) == prefix)
                atomicAdd(&hist[(key[j] >> (p * 8)) & 255u], 1u);
        __syncthreads();
        if (t == 0) {
            unsigned cum = 0;
            int d = 255;
            for (; d > 0; --d) { cum += hist[d]; if ((int)cum >= kk) break; }
            if ((int)cum < kk) cum += hist[0]; // d==0 fallthrough
            sh_sel = (unsigned)d;
            sh_kk  = kk - (int)(cum - hist[d]);
        }
        __syncthreads();
        prefix |= sh_sel << (p * 8);
        pmask  |= 0xFFu  << (p * 8);
        kk = sh_kk;
        __syncthreads();
    }
    const unsigned T      = prefix;  // 64th-largest key
    const int      budget = kk;      // how many ==T survive

    if (t == 0) { nsel = 0; neq = 0; }
    __syncthreads();

    // --- scatter: keep top-64, zero the rest, collect (idx,val) list --------
    #pragma unroll
    for (int j = 0; j < 24; ++j) {
        bool keep = false;
        if (key[j] > T) keep = true;
        else if (key[j] == T) { int e = atomicAdd(&neq, 1); keep = (e < budget); }
        __builtin_nontemporal_store(keep ? v[j] : 0.0f, &hrow[t + j * 256]);
        if (keep) {
            int s = atomicAdd(&nsel, 1);
            if (s < 80) { tval[s] = v[j]; tidx[s] = t + j * 256; }
        }
    }
    __syncthreads();
    int n = nsel < 80 ? nsel : 80;

    // --- deterministic summation order: odd-even sort by feature index ------
    for (int phase = 0; phase < 80; ++phase) {
        int i = 2 * t + (phase & 1);
        if (i + 1 < n) {
            if (tidx[i] > tidx[i + 1]) {
                int   ti = tidx[i];  tidx[i] = tidx[i + 1]; tidx[i + 1] = ti;
                float tv = tval[i];  tval[i] = tval[i + 1]; tval[i + 1] = tv;
            }
        }
        __syncthreads();
    }

    // --- sparse classifier GEMM: 250 float4 columns, gathers hit L2 ---------
    if (t < 250) {
        const int c = t * 4;
        v4f acc = *(const v4f*)(bc + c);
        for (int j = 0; j < n; ++j) {
            const float s = tval[j];
            const v4f w = *(const v4f*)(Wc + (size_t)tidx[j] * C_DIM + c);
            acc[0] += s * w[0];
            acc[1] += s * w[1];
            acc[2] += s * w[2];
            acc[3] += s * w[3];
        }
        __builtin_nontemporal_store(acc, (v4f*)(logits + (size_t)row * C_DIM + c));
    }
}

// ---------------------------------------------------------------------------
extern "C" void kernel_launch(void* const* d_in, const int* in_sizes, int n_in,
                              void* d_out, int out_size, void* d_ws, size_t ws_size,
                              hipStream_t stream) {
    (void)in_sizes; (void)n_in; (void)d_ws; (void)ws_size; (void)out_size;
    const float* x  = (const float*)d_in[0];   // [4096, 768]
    const float* Wi = (const float*)d_in[1];   // [768, 6144]
    const float* bi = (const float*)d_in[2];   // [6144]
    const float* Wc = (const float*)d_in[3];   // [6144, 1000]
    const float* bc = (const float*)d_in[4];   // [1000]
    // d_in[5] is k, fixed to 64 by the reference.

    float* sparse = (float*)d_out;                       // [4096*6144]
    float* logits = sparse + (size_t)B_DIM * F_DIM;      // [4096*1000]

    dim3 g1(F_DIM / TILE, B_DIM / TILE);  // (48, 32)
    gemm1_bias_relu<<<g1, dim3(256), 0, stream>>>(x, Wi, bi, sparse);
    topk_scatter_cls<<<dim3(B_DIM), dim3(256), 0, stream>>>(sparse, Wc, bc, logits);
}
